// GAT_70497593197184
// MI455X (gfx1250) — matrix operand order repaired and checked by reading, not compiled, not run on hardware
//
#include <hip/hip_runtime.h>

typedef __attribute__((ext_vector_type(16))) _Float16 v16h;
typedef __attribute__((ext_vector_type(8)))  _Float16 v8h;
typedef __attribute__((ext_vector_type(8)))  float    v8f;

#define N_NODES  10000
#define N_EDGES  160000
#define E_TOT    170000
#define N_GRAPHS 20
#define NEG_SLOPE 0.2f
#define GN_EPS    1e-5f
#define SM_EPS    1e-16f

// ---------------- conversion kernels ----------------

__global__ void k_cvt_a(const float* __restrict__ X, _Float16* __restrict__ A,
                        int Kin, int Kp, long tot) {
  long i = (long)blockIdx.x * blockDim.x + threadIdx.x;
  if (i >= tot) return;
  int m = (int)(i / Kp), k = (int)(i % Kp);
  A[i] = (k < Kin) ? (_Float16)X[(long)m * Kin + k] : (_Float16)0.0f;
}

// W: fp32 [Kin, Fout] row-major  ->  Wt: f16 [Np, Kp] (transposed, zero-padded)
__global__ void k_cvt_w(const float* __restrict__ W, _Float16* __restrict__ Wt,
                        int Kin, int Fout, int Kp, long tot) {
  long i = (long)blockIdx.x * blockDim.x + threadIdx.x;
  if (i >= tot) return;
  int n = (int)(i / Kp), k = (int)(i % Kp);
  Wt[i] = (n < Fout && k < Kin) ? (_Float16)W[(long)k * Fout + n] : (_Float16)0.0f;
}

// ---------------- WMMA GEMM:  C[M,Fout] = A[M,Kp] * Wt^T  ----------------
// Each wave computes a 16(M) x 64(N) strip: one A fragment per k-step is
// reused across 4 B fragments / 4 accumulators (4 WMMAs per k-step).
// 4 waves per block -> block covers 16 x 256 of C. A/B use contiguous k = 16*hi + s (same in both); C layout per
// CDNA5 ISA 7.12.2 (wave32, 16-bit A 16x32 / B 32x16 / f32 C 16x16).
__global__ void __launch_bounds__(128)
k_gemm_wmma(const _Float16* __restrict__ A, const _Float16* __restrict__ Wt,
            float* __restrict__ C, int Kp, int Fout, int ldc) {
  const int wave = threadIdx.x >> 5;
  const int lane = threadIdx.x & 31;
  const int hi   = lane >> 4;      // lane half: 0 = lanes 0-15, 1 = lanes 16-31
  const int r    = lane & 15;
  const int mBase = blockIdx.x * 16;
  const int nWave = blockIdx.y * 256 + wave * 64;

  const _Float16* ap = A + (long)(mBase + r) * Kp;            // A row m = mBase+r
  const _Float16* bp0 = Wt + (long)(nWave +  0 + r) * Kp;     // W cols (Wt rows)
  const _Float16* bp1 = Wt + (long)(nWave + 16 + r) * Kp;
  const _Float16* bp2 = Wt + (long)(nWave + 32 + r) * Kp;
  const _Float16* bp3 = Wt + (long)(nWave + 48 + r) * Kp;

  v8f acc0 = {}, acc1 = {}, acc2 = {}, acc3 = {};
  for (int k0 = 0; k0 < Kp; k0 += 32) {
    // A fragment: lane<16 holds K = k0+[0..16); lane>=16: k0+[16..32)
    v8h alo = *(const v8h*)(ap + k0 + 16 * hi);
    v8h ahi = *(const v8h*)(ap + k0 + 16 * hi + 8);
    v16h a;
#pragma unroll
    for (int j = 0; j < 8; ++j) { a[j] = alo[j]; a[j + 8] = ahi[j]; }
    // B fragments: lane<16 holds K = k0+[0..16); lane>=16: k0+[16..32)
    const int boff = k0 + 16 * hi;
    v16h b0 = *(const v16h*)(bp0 + boff);
    v16h b1 = *(const v16h*)(bp1 + boff);
    v16h b2 = *(const v16h*)(bp2 + boff);
    v16h b3 = *(const v16h*)(bp3 + boff);
    acc0 = __builtin_amdgcn_wmma_f32_16x16x32_f16(false, a, false, b0, (short)0, acc0, false, false);
    acc1 = __builtin_amdgcn_wmma_f32_16x16x32_f16(false, a, false, b1, (short)0, acc1, false, false);
    acc2 = __builtin_amdgcn_wmma_f32_16x16x32_f16(false, a, false, b2, (short)0, acc2, false, false);
    acc3 = __builtin_amdgcn_wmma_f32_16x16x32_f16(false, a, false, b3, (short)0, acc3, false, false);
  }
  // C/D layout: lane half selects M block of 8; VGPR v -> M = v + 8*hi, N = r
  v8f accs[4] = {acc0, acc1, acc2, acc3};
#pragma unroll
  for (int t = 0; t < 4; ++t) {
    const int n = nWave + t * 16 + r;
    if (n < Fout) {
#pragma unroll
      for (int v = 0; v < 8; ++v)
        C[(long)(mBase + v + 8 * hi) * ldc + n] = accs[t][v];
    }
  }
}

// ---------------- attention coefficients ----------------

__global__ void k_att(const float* __restrict__ Hm, const float* __restrict__ asv,
                      const float* __restrict__ adv, float* __restrict__ As,
                      float* __restrict__ Ad, int Hh, int Cc) {
  int i = blockIdx.x * blockDim.x + threadIdx.x;  // i = n*Hh + h
  if (i >= N_NODES * Hh) return;
  int n = i / Hh, h = i % Hh;
  const float* hp = Hm + (long)n * Hh * Cc + (long)h * Cc;
  const float* sp = asv + (long)h * Cc;
  const float* dp = adv + (long)h * Cc;
  float s = 0.f, d = 0.f;
  for (int c = 0; c < Cc; ++c) { float v = hp[c]; s += v * sp[c]; d += v * dp[c]; }
  As[i] = s; Ad[i] = d;
}

// ---------------- fills ----------------

__global__ void k_fill_u32(unsigned* p, unsigned v, long n) {
  long i = (long)blockIdx.x * blockDim.x + threadIdx.x;
  if (i < n) p[i] = v;
}
__global__ void k_fill_f32(float* p, float v, long n) {
  long i = (long)blockIdx.x * blockDim.x + threadIdx.x;
  if (i < n) p[i] = v;
}

// ---------------- edge softmax ----------------

__device__ __forceinline__ unsigned fmap_ord(float f) {
  unsigned u = __float_as_uint(f);
  return (u & 0x80000000u) ? ~u : (u | 0x80000000u);
}
__device__ __forceinline__ float funmap_ord(unsigned u) {
  return (u & 0x80000000u) ? __uint_as_float(u & 0x7fffffffu) : __uint_as_float(~u);
}

__global__ void k_edge_max(const int* __restrict__ ei, const float* __restrict__ As,
                           const float* __restrict__ Ad, unsigned* __restrict__ Emax,
                           int Hh) {
  int i = blockIdx.x * blockDim.x + threadIdx.x;  // i = e*Hh + h
  if (i >= E_TOT * Hh) return;
  int e = i / Hh, h = i % Hh;
  int s = (e < N_EDGES) ? ei[e] : (e - N_EDGES);
  int d = (e < N_EDGES) ? ei[N_EDGES + e] : (e - N_EDGES);
  float v = As[s * Hh + h] + Ad[d * Hh + h];
  v = (v > 0.f) ? v : NEG_SLOPE * v;
  atomicMax(&Emax[d * Hh + h], fmap_ord(v));
}

__global__ void k_edge_exp(const int* __restrict__ ei, const float* __restrict__ As,
                           const float* __restrict__ Ad, const unsigned* __restrict__ Emax,
                           float* __restrict__ Ex, float* __restrict__ Den, int Hh) {
  int i = blockIdx.x * blockDim.x + threadIdx.x;
  if (i >= E_TOT * Hh) return;
  int e = i / Hh, h = i % Hh;
  int s = (e < N_EDGES) ? ei[e] : (e - N_EDGES);
  int d = (e < N_EDGES) ? ei[N_EDGES + e] : (e - N_EDGES);
  float v = As[s * Hh + h] + Ad[d * Hh + h];
  v = (v > 0.f) ? v : NEG_SLOPE * v;
  float ex = expf(v - funmap_ord(Emax[d * Hh + h]));
  Ex[i] = ex;
  atomicAdd(&Den[d * Hh + h], ex);
}

// ---------------- scatter: out[dst] += h[src] * alpha ----------------

__global__ void k_init_bias(float* __restrict__ O, const float* __restrict__ b,
                            int Wd, long tot) {
  long i = (long)blockIdx.x * blockDim.x + threadIdx.x;
  if (i < tot) O[i] = b[(int)(i % Wd)];
}

__global__ void k_scatter(const int* __restrict__ ei, const float* __restrict__ Hm,
                          const float* __restrict__ Ex, const float* __restrict__ Den,
                          float* __restrict__ O, int Hh, int Cc) {
  int e = blockIdx.x;
  int s = (e < N_EDGES) ? ei[e] : (e - N_EDGES);
  int d = (e < N_EDGES) ? ei[N_EDGES + e] : (e - N_EDGES);
  int Wd = Hh * Cc;
  for (int f = threadIdx.x; f < Wd; f += blockDim.x) {
    int h = f / Cc;
    float alpha = Ex[(long)e * Hh + h] / (Den[d * Hh + h] + SM_EPS);
    atomicAdd(&O[(long)d * Wd + f], Hm[(long)s * Wd + f] * alpha);
  }
}

// ---------------- GraphNorm (+ELU) ----------------

__global__ void k_count(const int* __restrict__ batch, float* __restrict__ cnt) {
  int n = blockIdx.x * blockDim.x + threadIdx.x;
  if (n < N_NODES) atomicAdd(&cnt[batch[n]], 1.f);
}
__global__ void k_gn_sum(const float* __restrict__ X, const int* __restrict__ batch,
                         float* __restrict__ gsum, int Wd, long tot) {
  long i = (long)blockIdx.x * blockDim.x + threadIdx.x;
  if (i >= tot) return;
  int n = (int)(i / Wd), f = (int)(i % Wd);
  atomicAdd(&gsum[(long)batch[n] * Wd + f], X[i]);
}
__global__ void k_gn_ctr(float* __restrict__ X, const int* __restrict__ batch,
                         const float* __restrict__ gsum, const float* __restrict__ cnt,
                         const float* __restrict__ ga, float* __restrict__ gvar,
                         int Wd, long tot) {
  long i = (long)blockIdx.x * blockDim.x + threadIdx.x;
  if (i >= tot) return;
  int n = (int)(i / Wd), f = (int)(i % Wd);
  int g = batch[n];
  float mean = gsum[(long)g * Wd + f] / cnt[g];
  float c = X[i] - mean * ga[f];
  X[i] = c;
  atomicAdd(&gvar[(long)g * Wd + f], c * c);
}
__global__ void k_gn_fin(float* __restrict__ X, const int* __restrict__ batch,
                         const float* __restrict__ gvar, const float* __restrict__ cnt,
                         const float* __restrict__ gw, const float* __restrict__ gb,
                         int Wd, long tot) {
  long i = (long)blockIdx.x * blockDim.x + threadIdx.x;
  if (i >= tot) return;
  int n = (int)(i / Wd), f = (int)(i % Wd);
  int g = batch[n];
  float var = gvar[(long)g * Wd + f] / cnt[g];
  float y = gw[f] * X[i] * rsqrtf(var + GN_EPS) + gb[f];
  X[i] = (y > 0.f) ? y : expm1f(y);   // ELU
}

// ---------------- host orchestration ----------------

static inline long cdiv(long a, long b) { return (a + b - 1) / b; }

extern "C" void kernel_launch(void* const* d_in, const int* in_sizes, int n_in,
                              void* d_out, int out_size, void* d_ws, size_t ws_size,
                              hipStream_t stream) {
  (void)in_sizes; (void)n_in; (void)out_size; (void)ws_size;
  const float* x     = (const float*)d_in[0];
  const int*   ei    = (const int*)d_in[1];
  const int*   batch = (const int*)d_in[2];
  const float* Wm[4] = {(const float*)d_in[3], (const float*)d_in[7],
                        (const float*)d_in[11], (const float*)d_in[15]};
  const float* ASm[4]= {(const float*)d_in[4], (const float*)d_in[8],
                        (const float*)d_in[12], (const float*)d_in[16]};
  const float* ADm[4]= {(const float*)d_in[5], (const float*)d_in[9],
                        (const float*)d_in[13], (const float*)d_in[17]};
  const float* Bm[4] = {(const float*)d_in[6], (const float*)d_in[10],
                        (const float*)d_in[14], (const float*)d_in[18]};
  const float* GWm[3]= {(const float*)d_in[19], (const float*)d_in[22], (const float*)d_in[25]};
  const float* GBm[3]= {(const float*)d_in[20], (const float*)d_in[23], (const float*)d_in[26]};
  const float* GAm[3]= {(const float*)d_in[21], (const float*)d_in[24], (const float*)d_in[27]};

  // workspace carve-up (256B aligned)
  char* ws = (char*)d_ws;
  size_t off = 0;
  auto carve = [&](size_t bytes) {
    char* p = ws + off;
    off += (bytes + 255) & ~(size_t)255;
    return p;
  };
  _Float16* A16  = (_Float16*)carve((size_t)N_NODES * 1024 * sizeof(_Float16));
  _Float16* W16  = (_Float16*)carve((size_t)1024 * 1024 * sizeof(_Float16));
  float*    H32  = (float*)   carve((size_t)N_NODES * 1024 * sizeof(float));
  float*    O32  = (float*)   carve((size_t)N_NODES * 1024 * sizeof(float));
  float*    Asb  = (float*)   carve((size_t)N_NODES * 4 * sizeof(float));
  float*    Adb  = (float*)   carve((size_t)N_NODES * 4 * sizeof(float));
  unsigned* Emax = (unsigned*)carve((size_t)N_NODES * 4 * sizeof(unsigned));
  float*    Den  = (float*)   carve((size_t)N_NODES * 4 * sizeof(float));
  float*    Ex   = (float*)   carve((size_t)E_TOT * 4 * sizeof(float));
  float*    Gsum = (float*)   carve((size_t)N_GRAPHS * 1024 * sizeof(float));
  float*    Gvar = (float*)   carve((size_t)N_GRAPHS * 1024 * sizeof(float));
  float*    Gcnt = (float*)   carve((size_t)N_GRAPHS * sizeof(float));

  const int Hs[4]  = {4, 4, 4, 1};
  const int Cs[4]  = {256, 256, 256, 121};
  const int Kin[4] = {50, 1024, 1024, 1024};

  const float* Xin = x;
  for (int li = 0; li < 4; ++li) {
    const int Hh = Hs[li], Cc = Cs[li];
    const int Fout = Hh * Cc;
    const int K  = Kin[li];
    const int Kp = (K + 31) & ~31;        // 64 / 1024
    const int Np = (Fout + 255) & ~255;   // 1024 / 256 (wave strip = 64, block = 256)

    // 1) convert activation & weight to f16 (weight transposed + padded)
    long atot = (long)N_NODES * Kp;
    k_cvt_a<<<cdiv(atot, 256), 256, 0, stream>>>(Xin, A16, K, Kp, atot);
    long wtot = (long)Np * Kp;
    k_cvt_w<<<cdiv(wtot, 256), 256, 0, stream>>>(Wm[li], W16, K, Fout, Kp, wtot);

    // 2) GEMM via WMMA: H32[N, Fout] = A16 * W
    dim3 ggrid(N_NODES / 16, Np / 256);
    k_gemm_wmma<<<ggrid, 128, 0, stream>>>(A16, W16, H32, Kp, Fout, Fout);

    // 3) per-node attention coefficients
    int nh = N_NODES * Hh;
    k_att<<<cdiv(nh, 256), 256, 0, stream>>>(H32, ASm[li], ADm[li], Asb, Adb, Hh, Cc);

    // 4) edge softmax: max, exp, denominator
    k_fill_u32<<<cdiv(nh, 256), 256, 0, stream>>>(Emax, 0x007FFFFFu, nh); // fmap(-inf)
    k_fill_f32<<<cdiv(nh, 256), 256, 0, stream>>>(Den, 0.f, nh);
    long ehtot = (long)E_TOT * Hh;
    k_edge_max<<<cdiv(ehtot, 256), 256, 0, stream>>>(ei, Asb, Adb, Emax, Hh);
    k_edge_exp<<<cdiv(ehtot, 256), 256, 0, stream>>>(ei, Asb, Adb, Emax, Ex, Den, Hh);

    // 5) weighted scatter-sum (output pre-initialized with bias)
    float* Odst = (li == 3) ? (float*)d_out : O32;
    long otot = (long)N_NODES * Fout;
    k_init_bias<<<cdiv(otot, 256), 256, 0, stream>>>(Odst, Bm[li], Fout, otot);
    k_scatter<<<E_TOT, 256, 0, stream>>>(ei, H32, Ex, Den, Odst, Hh, Cc);

    // 6) GraphNorm + ELU (layers 1..3)
    if (li < 3) {
      long gt = (long)N_GRAPHS * Fout;
      k_fill_f32<<<cdiv(gt, 256), 256, 0, stream>>>(Gsum, 0.f, gt);
      k_fill_f32<<<cdiv(gt, 256), 256, 0, stream>>>(Gvar, 0.f, gt);
      k_fill_f32<<<1, 32, 0, stream>>>(Gcnt, 0.f, N_GRAPHS);
      k_count<<<cdiv(N_NODES, 256), 256, 0, stream>>>(batch, Gcnt);
      k_gn_sum<<<cdiv(otot, 256), 256, 0, stream>>>(O32, batch, Gsum, Fout, otot);
      k_gn_ctr<<<cdiv(otot, 256), 256, 0, stream>>>(O32, batch, Gsum, Gcnt, GAm[li],
                                                    Gvar, Fout, otot);
      k_gn_fin<<<cdiv(otot, 256), 256, 0, stream>>>(O32, batch, Gvar, Gcnt, GWm[li],
                                                    GBm[li], Fout, otot);
      Xin = O32;
    }
  }
}
